// FlowModel_87067577025353
// MI455X (gfx1250) — matrix operand order repaired and checked by reading, not compiled, not run on hardware
//
#include <hip/hip_runtime.h>
#include <hip/hip_bf16.h>

// ---------------------------------------------------------------------------
// Problem constants (match reference)
// ---------------------------------------------------------------------------
#define NB_B   32
#define NBB_   512
#define NN     (NB_B * NBB_)      // 16384 nodes
#define DEG_   16
#define EE     (NN * DEG_)        // 262144 edges
#define DD     64
#define LL     4

typedef __attribute__((ext_vector_type(16))) __bf16 v16bf;
typedef __attribute__((ext_vector_type(8)))  float  v8f;

// ---------------------------------------------------------------------------
// helpers
// ---------------------------------------------------------------------------
__device__ __forceinline__ unsigned int encF(float f) {
    unsigned int u = __float_as_uint(f);
    return (u & 0x80000000u) ? ~u : (u | 0x80000000u);
}
__device__ __forceinline__ float decF(unsigned int u) {
    unsigned int b = (u & 0x80000000u) ? (u & 0x7fffffffu) : ~u;
    return __uint_as_float(b);
}
__device__ __forceinline__ float gelu_tanh(float x) {
    float x3 = x * x * x;
    return 0.5f * x * (1.0f + tanhf(0.7978845608028654f * (x + 0.044715f * x3)));
}

// ---------------------------------------------------------------------------
// generic clears
// ---------------------------------------------------------------------------
__global__ void clear_f(float* p, int n) {
    int i = blockIdx.x * blockDim.x + threadIdx.x;
    if (i < n) p[i] = 0.0f;
}
__global__ void clear_u(unsigned int* p, int n) {
    int i = blockIdx.x * blockDim.x + threadIdx.x;
    if (i < n) p[i] = 0u;
}

// ---------------------------------------------------------------------------
// transpose (I, 64, 64) f32 [i][j][k]  ->  bf16 [i][k][j]  (WMMA-B friendly)
// ---------------------------------------------------------------------------
__global__ void transpose_bf16(const float* __restrict__ src, __bf16* __restrict__ dst, int I) {
    int idx = blockIdx.x * blockDim.x + threadIdx.x;
    if (idx >= I * 4096) return;
    int i = idx >> 12, rem = idx & 4095, k = rem >> 6, j = rem & 63;
    dst[i * 4096 + k * 64 + j] = (__bf16)src[i * 4096 + j * 64 + k];
}

// ---------------------------------------------------------------------------
// node embedding: a = [x @ W_bb | rot9 @ W_rotE]   (N x 64)
// ---------------------------------------------------------------------------
__global__ __launch_bounds__(64) void embed_node(const float* __restrict__ x,
                                                 const float* __restrict__ rot,
                                                 const float* __restrict__ Wbb,
                                                 const float* __restrict__ WrotE,
                                                 float* __restrict__ aF,
                                                 __bf16* __restrict__ aBf) {
    int n = blockIdx.x, k = threadIdx.x;
    float acc = 0.0f;
    if (k < 48) {
        const float* xr = x + n * 64;
        for (int i = 0; i < 64; ++i) acc += xr[i] * Wbb[i * 48 + k];
    } else {
        int kk = k - 48;
        const float* rr = rot + n * 9;
        for (int i = 0; i < 9; ++i) acc += rr[i] * WrotE[i * 16 + kk];
    }
    aF[n * 64 + k] = acc;
    aBf[n * 64 + k] = (__bf16)acc;
}

// ---------------------------------------------------------------------------
// WMMA bilinear: out[n,k] = sum_i a[n,i] * (A @ W[i])[n,k]
//   Abf: N x 64 bf16 ; Af: N x 64 f32 (scales) ; Wt: [i][k*64+j] bf16
// one 16-node tile per block, 4 waves = 4 k-tiles
// ---------------------------------------------------------------------------
__global__ __launch_bounds__(128) void bilinear_wmma(const __bf16* __restrict__ Abf,
                                                     const float* __restrict__ Af,
                                                     const __bf16* __restrict__ Wt,
                                                     float* __restrict__ outF,
                                                     __bf16* __restrict__ outBf) {
    __shared__ float sA[16 * 64];
    int tile = blockIdx.x;
    int wave = threadIdx.x >> 5;
    int lane = threadIdx.x & 31;
    for (int idx = threadIdx.x; idx < 16 * 64; idx += 128)
        sA[idx] = Af[tile * 1024 + idx];
    __syncthreads();

    int half = lane >> 4, lm = lane & 15;
    int c0 = half * 16;
    const __bf16* Arow = Abf + (tile * 16 + lm) * 64;
    v16bf a0, a1;
#pragma unroll
    for (int e = 0; e < 16; ++e) {
        a0[e]     = Arow[c0 + e];
        a1[e]     = Arow[32 + c0 + e];
    }
    int kt = wave;
    v8f acc = {};
    for (int i = 0; i < 64; ++i) {
        const __bf16* Brow = Wt + i * 4096 + (kt * 16 + lm) * 64 + half * 16;
        if (i < 63) __builtin_prefetch(Brow + 4096, 0, 0);
        v16bf b0, b1;
#pragma unroll
        for (int e = 0; e < 16; ++e) { b0[e] = Brow[e]; b1[e] = Brow[32 + e]; }
        v8f s = {};
        s = __builtin_amdgcn_wmma_f32_16x16x32_bf16(false, a0, false, b0, (short)0, s, false, false);
        s = __builtin_amdgcn_wmma_f32_16x16x32_bf16(false, a1, false, b1, (short)0, s, false, false);
#pragma unroll
        for (int r = 0; r < 8; ++r) acc[r] += sA[(r + half * 8) * 64 + i] * s[r];
    }
#pragma unroll
    for (int r = 0; r < 8; ++r) {
        int n = tile * 16 + r + half * 8;
        int k = kt * 16 + lm;
        outF[n * 64 + k] = acc[r];
        if (outBf) outBf[n * 64 + k] = (__bf16)acc[r];
    }
}

// ---------------------------------------------------------------------------
// WMMA GEMM: out = A(N x 64) @ B(64 x 64); B pre-transposed [k*64+j] bf16.
// perBatch != 0 -> B indexed by batch (tile/32)*4096
// ---------------------------------------------------------------------------
__global__ __launch_bounds__(128) void gemm64_wmma(const __bf16* __restrict__ Abf,
                                                   const __bf16* __restrict__ Bt,
                                                   float* __restrict__ outF,
                                                   __bf16* __restrict__ outBf,
                                                   int perBatch) {
    int tile = blockIdx.x;
    int wave = threadIdx.x >> 5;
    int lane = threadIdx.x & 31;
    int half = lane >> 4, lm = lane & 15;
    int c0 = half * 16;
    const __bf16* Arow = Abf + (tile * 16 + lm) * 64;
    v16bf a0, a1;
#pragma unroll
    for (int e = 0; e < 16; ++e) {
        a0[e]     = Arow[c0 + e];
        a1[e]     = Arow[32 + c0 + e];
    }
    const __bf16* Bbase = Bt + (perBatch ? (tile >> 5) * 4096 : 0);
    int kt = wave;
    const __bf16* Brow = Bbase + (kt * 16 + lm) * 64 + half * 16;
    v16bf b0, b1;
#pragma unroll
    for (int e = 0; e < 16; ++e) { b0[e] = Brow[e]; b1[e] = Brow[32 + e]; }
    v8f acc = {};
    acc = __builtin_amdgcn_wmma_f32_16x16x32_bf16(false, a0, false, b0, (short)0, acc, false, false);
    acc = __builtin_amdgcn_wmma_f32_16x16x32_bf16(false, a1, false, b1, (short)0, acc, false, false);
#pragma unroll
    for (int r = 0; r < 8; ++r) {
        int n = tile * 16 + r + half * 8;
        int k = kt * 16 + lm;
        outF[n * 64 + k] = acc[r];
        if (outBf) outBf[n * 64 + k] = (__bf16)acc[r];
    }
}

// ---------------------------------------------------------------------------
// lattice init: te = [sin(t*2^f), cos(t*2^f)] (16) ++ lat9 @ W_latE (48),
// then lat = bilinear(lv, lv, W_comb_lat)   (32 x 64)
// ---------------------------------------------------------------------------
__global__ __launch_bounds__(64) void lat_init(const float* __restrict__ t,
                                               const float* __restrict__ lattice,
                                               const float* __restrict__ WlatE,
                                               const float* __restrict__ Wcl,
                                               float* __restrict__ latF) {
    int b = blockIdx.x, k = threadIdx.x;
    __shared__ float lv[64];
    float v;
    if (k < 8)        v = sinf(t[b] * exp2f((float)k));
    else if (k < 16)  v = cosf(t[b] * exp2f((float)(k - 8)));
    else {
        float acc = 0.0f;
        for (int i = 0; i < 9; ++i) acc += lattice[b * 9 + i] * WlatE[i * 48 + (k - 16)];
        v = acc;
    }
    lv[k] = v;
    __syncthreads();
    float acc = 0.0f;
    for (int i = 0; i < 64; ++i) {
        float li = lv[i];
        const float* W = Wcl + i * 4096 + k;
        for (int j = 0; j < 64; ++j) acc += li * lv[j] * W[j * 64];
    }
    latF[b * 64 + k] = acc;
}

// ---------------------------------------------------------------------------
// edge pass 1: logits + segment max (order-preserving uint atomicMax)
// ---------------------------------------------------------------------------
__global__ __launch_bounds__(256) void edge_logits(const float* __restrict__ q,
                                                   const float* __restrict__ kk,
                                                   const int* __restrict__ src,
                                                   const int* __restrict__ dst,
                                                   float* __restrict__ logits,
                                                   unsigned int* __restrict__ mx) {
    int e = blockIdx.x * blockDim.x + threadIdx.x;
    if (e >= EE) return;
    int s = src[e], t = dst[e];
    const float* qr = q + t * 64;
    const float* kr = kk + s * 64;
    float acc = 0.0f;
#pragma unroll 8
    for (int i = 0; i < 64; ++i) acc += qr[i] * kr[i];
    acc *= 0.125f;  // 1/sqrt(64)
    logits[e] = acc;
    atomicMax(mx + t, encF(acc));
}

// edge pass 2: p = exp(logit - mx[dst]); den += p
__global__ __launch_bounds__(256) void edge_p(const float* __restrict__ logits,
                                              const int* __restrict__ dst,
                                              const unsigned int* __restrict__ mx,
                                              float* __restrict__ pbuf,
                                              float* __restrict__ den) {
    int e = blockIdx.x * blockDim.x + threadIdx.x;
    if (e >= EE) return;
    int t = dst[e];
    float p = expf(logits[e] - decF(mx[t]));
    pbuf[e] = p;
    atomicAdd(den + t, p);
}

// edge pass 3: alpha * (v[src] * we) scattered to upd[dst]
__global__ __launch_bounds__(256) void edge_upd(const float* __restrict__ pbuf,
                                                const float* __restrict__ den,
                                                const int* __restrict__ src,
                                                const int* __restrict__ dst,
                                                const float* __restrict__ trans,
                                                const float* __restrict__ vF,
                                                const float* __restrict__ W1,   // 16x32
                                                const float* __restrict__ W2,   // 32x64
                                                float* __restrict__ upd) {
    int e = blockIdx.x * blockDim.x + threadIdx.x;
    if (e >= EE) return;
    int s = src[e], t = dst[e];
    float alpha = pbuf[e] / (den[t] + 1e-9f);
    float dx = trans[s * 3 + 0] - trans[t * 3 + 0];
    float dy = trans[s * 3 + 1] - trans[t * 3 + 1];
    float dz = trans[s * 3 + 2] - trans[t * 3 + 2];
    float d = sqrtf(dx * dx + dy * dy + dz * dz);
    float rbf[16];
#pragma unroll
    for (int c = 0; c < 16; ++c) {
        float u = (d - (float)c * (5.0f / 15.0f)) * (1.0f / 0.3125f);
        rbf[c] = expf(-u * u);
    }
    float h[32];
#pragma unroll 4
    for (int o = 0; o < 32; ++o) {
        float x = 0.0f;
        for (int c = 0; c < 16; ++c) x += rbf[c] * W1[c * 32 + o];
        h[o] = gelu_tanh(x);
    }
    const float* vr = vF + s * 64;
    for (int k = 0; k < 64; ++k) {
        float w = 0.0f;
#pragma unroll 8
        for (int o = 0; o < 32; ++o) w += h[o] * W2[o * 64 + k];
        atomicAdd(upd + t * 64 + k, alpha * vr[k] * w);
    }
}

// node += upd (f32 + bf16 mirror)
__global__ __launch_bounds__(256) void add_upd(float* __restrict__ nodeF,
                                               __bf16* __restrict__ nodeBf,
                                               const float* __restrict__ upd) {
    int i = blockIdx.x * blockDim.x + threadIdx.x;
    if (i >= NN * 64) return;
    float v = nodeF[i] + upd[i];
    nodeF[i] = v;
    nodeBf[i] = (__bf16)v;
}

// ---------------------------------------------------------------------------
// lat' [b,k] = sum_ij lat[b,i] * node[last_b, j] * W_lat_inter[i,j,k]
// (reference computes this for all N and keeps row `last` — only 32 rows matter)
// ---------------------------------------------------------------------------
__global__ __launch_bounds__(64) void lat_inter(const float* __restrict__ latF,
                                                const float* __restrict__ nodeF,
                                                const float* __restrict__ Wli,
                                                float* __restrict__ lat2F) {
    int b = blockIdx.x, k = threadIdx.x;
    __shared__ float lv[64], nv[64];
    lv[k] = latF[b * 64 + k];
    nv[k] = nodeF[((b + 1) * NBB_ - 1) * 64 + k];
    __syncthreads();
    float acc = 0.0f;
    for (int i = 0; i < 64; ++i) {
        float li = lv[i];
        const float* W = Wli + i * 4096 + k;
        for (int j = 0; j < 64; ++j) acc += li * nv[j] * W[j * 64];
    }
    lat2F[b * 64 + k] = acc;
}

// M[b][k*64+j] = sum_i lat2[b,i] * W_node_inter[i,j,k]  (bf16, WMMA-B layout)
__global__ __launch_bounds__(256) void m_prep(const float* __restrict__ lat2F,
                                              const float* __restrict__ Wni,
                                              __bf16* __restrict__ mBf) {
    int b = blockIdx.x;
    __shared__ float lv[64];
    if (threadIdx.x < 64) lv[threadIdx.x] = lat2F[b * 64 + threadIdx.x];
    __syncthreads();
    for (int idx = threadIdx.x; idx < 4096; idx += 256) {
        int j = idx >> 6, k = idx & 63;
        float acc = 0.0f;
        for (int i = 0; i < 64; ++i) acc += lv[i] * Wni[i * 4096 + j * 64 + k];
        mBf[b * 4096 + k * 64 + j] = (__bf16)acc;
    }
}

// ---------------------------------------------------------------------------
// batch norm over N rows (per-feature) : stats then apply
// ---------------------------------------------------------------------------
__global__ __launch_bounds__(256) void bn_stats(const float* __restrict__ x,
                                                float* __restrict__ stats) {
    int f = blockIdx.x;  // 0..63
    __shared__ float s1[256], s2[256];
    float a = 0.0f, b = 0.0f;
    for (int r = threadIdx.x; r < NN; r += 256) {
        float v = x[r * 64 + f];
        a += v; b += v * v;
    }
    s1[threadIdx.x] = a; s2[threadIdx.x] = b;
    __syncthreads();
    for (int off = 128; off > 0; off >>= 1) {
        if (threadIdx.x < off) {
            s1[threadIdx.x] += s1[threadIdx.x + off];
            s2[threadIdx.x] += s2[threadIdx.x + off];
        }
        __syncthreads();
    }
    if (threadIdx.x == 0) {
        float m = s1[0] / (float)NN;
        float var = s2[0] / (float)NN - m * m;
        stats[f] = m;
        stats[64 + f] = rsqrtf(var + 1e-5f);
    }
}

__global__ __launch_bounds__(256) void bn_apply(const float* __restrict__ xin,
                                                const float* __restrict__ stats,
                                                const float* __restrict__ g,
                                                const float* __restrict__ b,
                                                float* __restrict__ outF,
                                                __bf16* __restrict__ outBf) {
    int i = blockIdx.x * blockDim.x + threadIdx.x;
    if (i >= NN * 64) return;
    int f = i & 63;
    float v = (xin[i] - stats[f]) * stats[64 + f] * g[f] + b[f];
    outF[i] = v;
    outBf[i] = (__bf16)v;
}

// batch norm of lattice features over B=32 rows
__global__ __launch_bounds__(64) void bn_lat(const float* __restrict__ lat2F,
                                             const float* __restrict__ g,
                                             const float* __restrict__ bta,
                                             float* __restrict__ latF) {
    int k = threadIdx.x;
    float m = 0.0f, q = 0.0f;
    for (int b = 0; b < NB_B; ++b) { float v = lat2F[b * 64 + k]; m += v; q += v * v; }
    m *= (1.0f / NB_B);
    float var = q * (1.0f / NB_B) - m * m;
    float inv = rsqrtf(var + 1e-5f);
    for (int b = 0; b < NB_B; ++b)
        latF[b * 64 + k] = (lat2F[b * 64 + k] - m) * inv * g[k] + bta[k];
}

// ---------------------------------------------------------------------------
// readout: pred_trans + pred_rotmats (Rodrigues)
// ---------------------------------------------------------------------------
__global__ __launch_bounds__(128) void readout_node(const float* __restrict__ nodeF,
                                                    const float* __restrict__ trans,
                                                    const float* __restrict__ Wt,   // 64x64x3
                                                    const float* __restrict__ Wr,   // 64x64x3
                                                    float* __restrict__ outTrans,
                                                    float* __restrict__ outRot) {
    int n = blockIdx.x * blockDim.x + threadIdx.x;
    if (n >= NN) return;
    float a[64];
    for (int i = 0; i < 64; ++i) a[i] = nodeF[n * 64 + i];
    float t0 = 0, t1 = 0, t2 = 0, r0 = 0, r1 = 0, r2 = 0;
    for (int i = 0; i < 64; ++i) {
        float ai = a[i];
        const float* wt = Wt + i * 192;
        const float* wr = Wr + i * 192;
        for (int j = 0; j < 64; ++j) {
            float s = ai * a[j];
            t0 += s * wt[j * 3 + 0]; t1 += s * wt[j * 3 + 1]; t2 += s * wt[j * 3 + 2];
            r0 += s * wr[j * 3 + 0]; r1 += s * wr[j * 3 + 1]; r2 += s * wr[j * 3 + 2];
        }
    }
    outTrans[n * 3 + 0] = trans[n * 3 + 0] + t0;
    outTrans[n * 3 + 1] = trans[n * 3 + 1] + t1;
    outTrans[n * 3 + 2] = trans[n * 3 + 2] + t2;

    float theta = sqrtf(r0 * r0 + r1 * r1 + r2 * r2) + 1e-8f;
    float kx = r0 / theta, ky = r1 / theta, kz = r2 / theta;
    float s = sinf(theta), c = cosf(theta), oc = 1.0f - c;
    float* R = outRot + n * 9;
    R[0] = 1.0f - oc * (ky * ky + kz * kz);
    R[1] = -s * kz + oc * kx * ky;
    R[2] =  s * ky + oc * kx * kz;
    R[3] =  s * kz + oc * kx * ky;
    R[4] = 1.0f - oc * (kx * kx + kz * kz);
    R[5] = -s * kx + oc * ky * kz;
    R[6] = -s * ky + oc * kx * kz;
    R[7] =  s * kx + oc * ky * kz;
    R[8] = 1.0f - oc * (kx * kx + ky * ky);
}

// pred_lattice[b, c] = sum_ij lat[b,i] lat[b,j] W_lat_read[i,j,c]
__global__ __launch_bounds__(64) void readout_lat(const float* __restrict__ latF,
                                                  const float* __restrict__ Wlr,
                                                  float* __restrict__ out) {
    int b = blockIdx.x;
    __shared__ float lv[64];
    lv[threadIdx.x] = latF[b * 64 + threadIdx.x];
    __syncthreads();
    int c = threadIdx.x;
    if (c >= 9) return;
    float acc = 0.0f;
    for (int i = 0; i < 64; ++i) {
        float li = lv[i];
        for (int j = 0; j < 64; ++j) acc += li * lv[j] * Wlr[i * 576 + j * 9 + c];
    }
    out[b * 9 + c] = acc;
}

// ---------------------------------------------------------------------------
// host orchestration
// ---------------------------------------------------------------------------
extern "C" void kernel_launch(void* const* d_in, const int* in_sizes, int n_in,
                              void* d_out, int out_size, void* d_ws, size_t ws_size,
                              hipStream_t stream) {
    const float* x        = (const float*)d_in[0];
    const float* trans_t  = (const float*)d_in[1];
    const float* rotmats  = (const float*)d_in[2];
    const float* t_in     = (const float*)d_in[3];
    const float* lattice  = (const float*)d_in[4];
    const float* W_bb     = (const float*)d_in[5];
    const float* W_rotE   = (const float*)d_in[6];
    const float* W_latE   = (const float*)d_in[7];
    const float* W_comb_node = (const float*)d_in[8];
    const float* W_comb_lat  = (const float*)d_in[9];
    const float* W_lat_inter = (const float*)d_in[10];
    const float* W_node_inter= (const float*)d_in[11];
    const float* Wq       = (const float*)d_in[12];
    const float* Wk       = (const float*)d_in[13];
    const float* Wv       = (const float*)d_in[14];
    const float* W_rbf1   = (const float*)d_in[15];
    const float* W_rbf2   = (const float*)d_in[16];
    const float* bn_ng    = (const float*)d_in[17];
    const float* bn_nb    = (const float*)d_in[18];
    const float* bn_lg    = (const float*)d_in[19];
    const float* bn_lb    = (const float*)d_in[20];
    const float* W_trans  = (const float*)d_in[21];
    const float* W_rot_rd = (const float*)d_in[22];
    const float* W_lat_rd = (const float*)d_in[23];
    const int*   src      = (const int*)d_in[25];
    const int*   dst      = (const int*)d_in[26];

    float* out = (float*)d_out;
    float* outTrans = out;                 // N*3
    float* outRot   = out + NN * 3;        // N*9
    float* outLat   = out + NN * 3 + NN * 9; // B*9

    // scratch carve-out
    char* wp = (char*)d_ws;
    auto carve = [&](size_t bytes) -> void* {
        void* p = (void*)wp;
        wp += (bytes + 255) & ~(size_t)255;
        return p;
    };
    float*  aF      = (float*)carve((size_t)NN * 64 * 4);
    __bf16* aBf     = (__bf16*)carve((size_t)NN * 64 * 2);
    float*  nodeF   = (float*)carve((size_t)NN * 64 * 4);
    __bf16* nodeBf  = (__bf16*)carve((size_t)NN * 64 * 2);
    float*  tmpF    = (float*)carve((size_t)NN * 64 * 4);
    float*  updF    = (float*)carve((size_t)NN * 64 * 4);
    float*  qF      = (float*)carve((size_t)NN * 64 * 4);
    float*  kF      = (float*)carve((size_t)NN * 64 * 4);
    float*  vF      = (float*)carve((size_t)NN * 64 * 4);
    float*  latF    = (float*)carve((size_t)NB_B * 64 * 4);
    float*  lat2F   = (float*)carve((size_t)NB_B * 64 * 4);
    __bf16* wtComb  = (__bf16*)carve((size_t)64 * 4096 * 2);
    __bf16* wtQ     = (__bf16*)carve((size_t)LL * 4096 * 2);
    __bf16* wtK     = (__bf16*)carve((size_t)LL * 4096 * 2);
    __bf16* wtV     = (__bf16*)carve((size_t)LL * 4096 * 2);
    __bf16* mBf     = (__bf16*)carve((size_t)NB_B * 4096 * 2);
    float*  logits  = (float*)carve((size_t)EE * 4);
    float*  pbuf    = (float*)carve((size_t)EE * 4);
    unsigned int* mx = (unsigned int*)carve((size_t)NN * 4);
    float*  den     = (float*)carve((size_t)NN * 4);
    float*  stats   = (float*)carve((size_t)128 * 4);

    const int TILES = NN / 16;  // 1024

    // --- weight transposes to WMMA-B bf16 layout ---
    transpose_bf16<<<(64 * 4096 + 255) / 256, 256, 0, stream>>>(W_comb_node, wtComb, 64);
    transpose_bf16<<<(LL * 4096 + 255) / 256, 256, 0, stream>>>(Wq, wtQ, LL);
    transpose_bf16<<<(LL * 4096 + 255) / 256, 256, 0, stream>>>(Wk, wtK, LL);
    transpose_bf16<<<(LL * 4096 + 255) / 256, 256, 0, stream>>>(Wv, wtV, LL);

    // --- embeddings & initial bilinears ---
    embed_node<<<NN, 64, 0, stream>>>(x, rotmats, W_bb, W_rotE, aF, aBf);
    bilinear_wmma<<<TILES, 128, 0, stream>>>(aBf, aF, wtComb, nodeF, nodeBf);
    lat_init<<<NB_B, 64, 0, stream>>>(t_in, lattice, W_latE, W_comb_lat, latF);

    // --- layers ---
    for (int l = 0; l < LL; ++l) {
        gemm64_wmma<<<TILES, 128, 0, stream>>>(nodeBf, wtQ + l * 4096, qF, nullptr, 0);
        gemm64_wmma<<<TILES, 128, 0, stream>>>(nodeBf, wtK + l * 4096, kF, nullptr, 0);
        gemm64_wmma<<<TILES, 128, 0, stream>>>(nodeBf, wtV + l * 4096, vF, nullptr, 0);

        clear_u<<<(NN + 255) / 256, 256, 0, stream>>>(mx, NN);
        clear_f<<<(NN + 255) / 256, 256, 0, stream>>>(den, NN);
        clear_f<<<(NN * 64 + 255) / 256, 256, 0, stream>>>(updF, NN * 64);

        edge_logits<<<(EE + 255) / 256, 256, 0, stream>>>(qF, kF, src, dst, logits, mx);
        edge_p<<<(EE + 255) / 256, 256, 0, stream>>>(logits, dst, mx, pbuf, den);
        edge_upd<<<(EE + 255) / 256, 256, 0, stream>>>(pbuf, den, src, dst, trans_t, vF,
                                                       W_rbf1 + l * 16 * 32,
                                                       W_rbf2 + l * 32 * 64, updF);
        add_upd<<<(NN * 64 + 255) / 256, 256, 0, stream>>>(nodeF, nodeBf, updF);

        // lattice interaction (only the 32 surviving rows), then node interaction
        lat_inter<<<NB_B, 64, 0, stream>>>(latF, nodeF, W_lat_inter, lat2F);
        m_prep<<<NB_B, 256, 0, stream>>>(lat2F, W_node_inter, mBf);
        gemm64_wmma<<<TILES, 128, 0, stream>>>(nodeBf, mBf, tmpF, nullptr, 1);

        bn_stats<<<64, 256, 0, stream>>>(tmpF, stats);
        bn_apply<<<(NN * 64 + 255) / 256, 256, 0, stream>>>(tmpF, stats, bn_ng, bn_nb,
                                                            nodeF, nodeBf);
        bn_lat<<<1, 64, 0, stream>>>(lat2F, bn_lg, bn_lb, latF);
    }

    // --- readouts ---
    readout_node<<<(NN + 127) / 128, 128, 0, stream>>>(nodeF, trans_t, W_trans, W_rot_rd,
                                                       outTrans, outRot);
    readout_lat<<<NB_B, 64, 0, stream>>>(latF, W_lat_rd, outLat);

    (void)in_sizes; (void)n_in; (void)out_size; (void)ws_size;
}
